// LigerLlama3GatedLinearAttention_78881369359032
// MI455X (gfx1250) — matrix operand order repaired and checked by reading, not compiled, not run on hardware
//
#include <hip/hip_runtime.h>

#define BB 2
#define SS 2048
#define HHID 2048
#define NHEAD 32
#define NKVH 8
#define HDIM 64
#define NGRP 4
#define MROWS (BB * SS)      // 4096
#define DQ (NHEAD * HDIM)    // 2048
#define DKV (NKVH * HDIM)    // 512

typedef __bf16 bf16;
typedef bf16  v8bf  __attribute__((ext_vector_type(8)));
typedef bf16  v16bf __attribute__((ext_vector_type(16)));
typedef float v8f   __attribute__((ext_vector_type(8)));
typedef unsigned int u32x4 __attribute__((ext_vector_type(4)));
typedef int i32x4 __attribute__((ext_vector_type(4)));
typedef int i32x8 __attribute__((ext_vector_type(8)));

__device__ __forceinline__ v8f vzero8() {
  v8f z;
#pragma unroll
  for (int i = 0; i < 8; ++i) z[i] = 0.0f;
  return z;
}

__device__ __forceinline__ v16bf cat16(v8bf lo, v8bf hi) {
  return __builtin_shufflevector(lo, hi, 0,1,2,3,4,5,6,7,8,9,10,11,12,13,14,15);
}

// A fragment (16x32 bf16): row fixed per lane, K split per ISA layout.
__device__ __forceinline__ v16bf fragA(const bf16* rowp, int kk, int lane) {
  int akb = (lane & 16) ? 8 : 0;
  v8bf lo = *(const v8bf*)(rowp + kk + akb);
  v8bf hi = *(const v8bf*)(rowp + kk + 16 + akb);
  return cat16(lo, hi);
}

// B fragment (32x16 bf16): column fixed per lane, K split per ISA layout.
__device__ __forceinline__ v16bf fragB(const bf16* colp, int kk, int lane) {
  int bkb = (lane & 16) ? 8 : 0;
  v8bf lo = *(const v8bf*)(colp + kk + bkb);
  v8bf hi = *(const v8bf*)(colp + kk + 16 + bkb);
  return cat16(lo, hi);
}

#define WMMA_BF(a, b, c) \
  __builtin_amdgcn_wmma_f32_16x16x32_bf16(false, (a), false, (b), (short)0, (c), false, false)

// ---------------------------------------------------------------------------
// TDM: 2D tile load global -> LDS via Tensor Data Mover.
// Descriptor per CDNA5 ISA (tile-start addressing:
// tensor dims == tile dims, real row stride in tensor_dim0_stride).
// 6-arg builtin form (clang-23 / therock headers).
// ---------------------------------------------------------------------------
__device__ __forceinline__ unsigned lds_off32(const void* p) {
  return (unsigned)(uintptr_t)p;
}

__device__ __forceinline__ void tdm_load_2d(unsigned lds_addr, const void* gaddr,
                                            unsigned tile_d0_elems, unsigned tile_d1_rows,
                                            unsigned row_stride_elems) {
  unsigned long long ga = (unsigned long long)(uintptr_t)gaddr;
  u32x4 g0;
  g0[0] = 1u;                                                 // count=1 (valid), user mode
  g0[1] = lds_addr;                                           // lds_addr (bytes)
  g0[2] = (unsigned)(ga & 0xFFFFFFFFu);                       // global_addr[31:0]
  g0[3] = (unsigned)((ga >> 32) & 0x01FFFFFFu) | (2u << 30);  // global_addr[56:32] | type=2
  i32x8 g1;
  g1[0] = (int)(1u << 16);                                    // data_size=1 -> 2 bytes
  g1[1] = (int)((tile_d0_elems & 0xFFFFu) << 16);             // tensor_dim0[15:0]
  g1[2] = (int)((tile_d0_elems >> 16) |
                ((tile_d1_rows & 0xFFFFu) << 16));            // tensor_dim0[31:16]|tensor_dim1[15:0]
  g1[3] = (int)((tile_d1_rows >> 16) |
                ((tile_d0_elems & 0xFFFFu) << 16));           // tensor_dim1[31:16]|tile_dim0
  g1[4] = (int)(tile_d1_rows & 0xFFFFu);                      // tile_dim1 (tile_dim2=0)
  g1[5] = (int)row_stride_elems;                              // tensor_dim0_stride[31:0]
  g1[6] = 0;                                                  // stride hi | dim1_stride lo
  g1[7] = 0;
  i32x4 z4;
  z4[0] = 0; z4[1] = 0; z4[2] = 0; z4[3] = 0;
  i32x8 z8;
#pragma unroll
  for (int i = 0; i < 8; ++i) z8[i] = 0;
  __builtin_amdgcn_tensor_load_to_lds(g0, g1, z4, z4, z8, 0);
}

// ---------------------------------------------------------------------------
// f32 -> bf16 convert
// ---------------------------------------------------------------------------
__global__ __launch_bounds__(256) void cvt_f32_bf16(const float* __restrict__ s,
                                                    bf16* __restrict__ d, size_t n) {
  size_t i = (size_t)blockIdx.x * blockDim.x + threadIdx.x;
  if (i < n) d[i] = (bf16)s[i];
}

// ---------------------------------------------------------------------------
// GEMM: Y[M,N] = X[M,K] * W[N,K]^T   (bf16 in, f32 out)
// Macro tile 128x64, 8 waves as 4x2, each wave 32x32 (4 WMMA per K step).
// Double-buffered LDS fed by TDM tensor_load_to_lds, waits on TENSORcnt.
// ---------------------------------------------------------------------------
__global__ __launch_bounds__(256) void gemm_bf16_wmma(const bf16* __restrict__ X,
                                                      const bf16* __restrict__ W,
                                                      float* __restrict__ Y,
                                                      int Kdim, int Ndim) {
  __shared__ __attribute__((aligned(16))) bf16 As[2][128][32];
  __shared__ __attribute__((aligned(16))) bf16 Bs[2][64][32];
  int tid = threadIdx.x, lane = tid & 31, wave = tid >> 5;
  int wm = wave & 3, wn = wave >> 2;
  int row0 = blockIdx.x * 128, col0 = blockIdx.y * 64;
  v8f c00 = vzero8(), c01 = vzero8(), c10 = vzero8(), c11 = vzero8();
  int nsteps = Kdim / 32;

  __builtin_prefetch(Y + (size_t)row0 * Ndim + col0, 0, 0);

  if (tid == 0) {
    tdm_load_2d(lds_off32(&As[0][0][0]), X + (size_t)row0 * Kdim, 32u, 128u, (unsigned)Kdim);
    tdm_load_2d(lds_off32(&Bs[0][0][0]), W + (size_t)col0 * Kdim, 32u, 64u, (unsigned)Kdim);
  }

  for (int s = 0; s < nsteps; ++s) {
    int buf = s & 1;
    if (tid == 0) {
      if (s + 1 < nsteps) {
        int k1 = (s + 1) * 32;
        tdm_load_2d(lds_off32(&As[buf ^ 1][0][0]), X + (size_t)row0 * Kdim + k1, 32u, 128u,
                    (unsigned)Kdim);
        tdm_load_2d(lds_off32(&Bs[buf ^ 1][0][0]), W + (size_t)col0 * Kdim + k1, 32u, 64u,
                    (unsigned)Kdim);
        __builtin_amdgcn_s_wait_tensorcnt((short)2);  // current buffer's 2 loads done
      } else {
        __builtin_amdgcn_s_wait_tensorcnt((short)0);
      }
    }
    __syncthreads();
    v16bf fa0 = fragA(&As[buf][wm * 32 + (lane & 15)][0], 0, lane);
    v16bf fa1 = fragA(&As[buf][wm * 32 + 16 + (lane & 15)][0], 0, lane);
    v16bf fb0 = fragB(&Bs[buf][wn * 32 + (lane & 15)][0], 0, lane);
    v16bf fb1 = fragB(&Bs[buf][wn * 32 + 16 + (lane & 15)][0], 0, lane);
    c00 = WMMA_BF(fa0, fb0, c00);
    c01 = WMMA_BF(fa0, fb1, c01);
    c10 = WMMA_BF(fa1, fb0, c10);
    c11 = WMMA_BF(fa1, fb1, c11);
    __syncthreads();
  }

  int hi8 = (lane & 16) >> 1;
  int colb = lane & 15;
  int orow = row0 + wm * 32 + hi8;
  int ocol = col0 + wn * 32 + colb;
#pragma unroll
  for (int r = 0; r < 8; r++) {
    Y[(size_t)(orow + r) * Ndim + ocol]           = c00[r];
    Y[(size_t)(orow + r) * Ndim + ocol + 16]      = c01[r];
    Y[(size_t)(orow + 16 + r) * Ndim + ocol]      = c10[r];
    Y[(size_t)(orow + 16 + r) * Ndim + ocol + 16] = c11[r];
  }
}

// ---------------------------------------------------------------------------
// RoPE + convert + head-major repack:  [B,S,nh,HD] f32 -> [B,nh,S,HD] bf16
// ---------------------------------------------------------------------------
__global__ __launch_bounds__(256) void rope_pack(const float* __restrict__ Qf,
                                                 bf16* __restrict__ Qb, int nheads) {
  size_t i = (size_t)blockIdx.x * blockDim.x + threadIdx.x;
  size_t total = (size_t)BB * nheads * SS * 32;
  if (i >= total) return;
  int j = (int)(i & 31);
  int s = (int)((i >> 5) & (SS - 1));
  int hb = (int)(i >> 16);
  int h = hb % nheads, b = hb / nheads;
  float freq = __expf(-((float)(2 * j) / 64.0f) * __logf(500000.0f));
  float ang = (float)s * freq;
  float sn, cs;
  __sincosf(ang, &sn, &cs);
  const float* src = Qf + (size_t)(b * SS + s) * (nheads * HDIM) + h * HDIM;
  float x1 = src[j], x2 = src[j + 32];
  bf16* dst = Qb + ((size_t)(b * nheads + h) * SS + s) * HDIM;
  dst[j]      = (bf16)(x1 * cs - x2 * sn);
  dst[j + 32] = (bf16)(x2 * cs + x1 * sn);
}

// [B,S,NKV,HD] f32 -> [B,NKV,S,HD] bf16
__global__ __launch_bounds__(256) void headmajor_cvt(const float* __restrict__ Vf,
                                                     bf16* __restrict__ Vb) {
  size_t i = (size_t)blockIdx.x * blockDim.x + threadIdx.x;
  size_t total = (size_t)BB * NKVH * SS * HDIM;
  if (i >= total) return;
  int d = (int)(i & 63);
  int s = (int)((i >> 6) & 2047);
  int h = (int)((i >> 17) & 7);
  int b = (int)(i >> 20);
  Vb[i] = (bf16)Vf[(size_t)(b * SS + s) * DKV + h * HDIM + d];
}

// ---------------------------------------------------------------------------
// Flash attention (teacher): grid (S/64, NH, B), block 256 (8 waves)
// ---------------------------------------------------------------------------
__global__ __launch_bounds__(256) void flash_attn(const bf16* __restrict__ Q,
                                                  const bf16* __restrict__ K,
                                                  const bf16* __restrict__ V,
                                                  float* __restrict__ O) {
  __shared__ __attribute__((aligned(16))) bf16 Qs[64][64];
  __shared__ __attribute__((aligned(16))) bf16 Ks[64][64];
  __shared__ __attribute__((aligned(16))) bf16 Vt[64][64];  // [v][tk]
  __shared__ __attribute__((aligned(16))) bf16 Ps[64][64];  // [tq][tk]
  __shared__ float aLds[64];
  __shared__ float lLds[64];

  int qb = blockIdx.x, h = blockIdx.y, b = blockIdx.z;
  int kvh = h >> 2;
  int tid = threadIdx.x, lane = tid & 31, wave = tid >> 5;
  int hi8 = (lane & 16) >> 1;
  int col = lane & 15;
  const bf16* Qp = Q + ((size_t)(b * NHEAD + h) * SS + qb * 64) * HDIM;
  const bf16* Kp = K + ((size_t)(b * NKVH + kvh) * SS) * HDIM;
  const bf16* Vp = V + ((size_t)(b * NKVH + kvh) * SS) * HDIM;

  int lr = tid >> 2, lc0 = (tid & 3) * 8;
#pragma unroll
  for (int cc = 0; cc < 64; cc += 32)
    *(v8bf*)&Qs[lr][lc0 + cc] = *(const v8bf*)(Qp + (size_t)lr * HDIM + lc0 + cc);

  float mreg[8], lreg[8];
#pragma unroll
  for (int r = 0; r < 8; r++) { mreg[r] = -3e38f; lreg[r] = 0.0f; }
  v8f oc0 = vzero8(), oc1 = vzero8();
  int qs = wave >> 1;
  int dt0 = (wave & 1) * 2;

  for (int kb = 0; kb <= qb; kb++) {
    __syncthreads();
#pragma unroll
    for (int cc = 0; cc < 64; cc += 32) {
      *(v8bf*)&Ks[lr][lc0 + cc] =
          *(const v8bf*)(Kp + (size_t)(kb * 64 + lr) * HDIM + lc0 + cc);
      v8bf vv = *(const v8bf*)(Vp + (size_t)(kb * 64 + lr) * HDIM + lc0 + cc);
#pragma unroll
      for (int j = 0; j < 8; j++) Vt[lc0 + cc + j][lr] = vv[j];
    }
    __syncthreads();
    if (wave < 4) {
      v8f sc0 = vzero8(), sc1 = vzero8(), sc2 = vzero8(), sc3 = vzero8();
      const bf16* arow = &Qs[wave * 16 + col][0];
#pragma unroll
      for (int kk = 0; kk < 64; kk += 32) {
        v16bf a = fragA(arow, kk, lane);
        sc0 = WMMA_BF(a, fragB(&Ks[col][0], kk, lane), sc0);
        sc1 = WMMA_BF(a, fragB(&Ks[16 + col][0], kk, lane), sc1);
        sc2 = WMMA_BF(a, fragB(&Ks[32 + col][0], kk, lane), sc2);
        sc3 = WMMA_BF(a, fragB(&Ks[48 + col][0], kk, lane), sc3);
      }
#pragma unroll
      for (int r = 0; r < 8; r++) {
        int rl = wave * 16 + r + hi8;
        float s0 = sc0[r] * 0.125f, s1 = sc1[r] * 0.125f;
        float s2 = sc2[r] * 0.125f, s3 = sc3[r] * 0.125f;
        if (kb == qb) {
          if (col > rl)      s0 = -3e38f;
          if (16 + col > rl) s1 = -3e38f;
          if (32 + col > rl) s2 = -3e38f;
          if (48 + col > rl) s3 = -3e38f;
        }
        float rm = fmaxf(fmaxf(s0, s1), fmaxf(s2, s3));
#pragma unroll
        for (int m = 1; m <= 8; m <<= 1) rm = fmaxf(rm, __shfl_xor(rm, m, 32));
        float mo = mreg[r];
        float mn = fmaxf(mo, rm);
        float al = __expf(mo - mn);
        float p0 = __expf(s0 - mn), p1 = __expf(s1 - mn);
        float p2 = __expf(s2 - mn), p3 = __expf(s3 - mn);
        float rs = p0 + p1 + p2 + p3;
#pragma unroll
        for (int m = 1; m <= 8; m <<= 1) rs += __shfl_xor(rs, m, 32);
        lreg[r] = al * lreg[r] + rs;
        mreg[r] = mn;
        Ps[rl][col]      = (bf16)p0;
        Ps[rl][16 + col] = (bf16)p1;
        Ps[rl][32 + col] = (bf16)p2;
        Ps[rl][48 + col] = (bf16)p3;
        if (col == 0) aLds[rl] = al;
      }
    }
    __syncthreads();
    {
      const bf16* arow = &Ps[qs * 16 + col][0];
#pragma unroll
      for (int r = 0; r < 8; r++) {
        float al = aLds[qs * 16 + r + hi8];
        oc0[r] *= al;
        oc1[r] *= al;
      }
#pragma unroll
      for (int kk = 0; kk < 64; kk += 32) {
        v16bf a = fragA(arow, kk, lane);
        oc0 = WMMA_BF(a, fragB(&Vt[dt0 * 16 + col][0], kk, lane), oc0);
        oc1 = WMMA_BF(a, fragB(&Vt[(dt0 + 1) * 16 + col][0], kk, lane), oc1);
      }
    }
  }
  __syncthreads();
  if (wave < 4 && col == 0) {
#pragma unroll
    for (int r = 0; r < 8; r++) lLds[wave * 16 + r + hi8] = lreg[r];
  }
  __syncthreads();
#pragma unroll
  for (int r = 0; r < 8; r++) {
    int rl = qs * 16 + r + hi8;
    float inv = 1.0f / lLds[rl];
    float* op = O + ((size_t)(b * SS + qb * 64 + rl) * NHEAD + h) * HDIM;
    op[dt0 * 16 + col]       = oc0[r] * inv;
    op[(dt0 + 1) * 16 + col] = oc1[r] * inv;
  }
}

// ---------------------------------------------------------------------------
// Student prep: rowwise softmax over 64, gate = logsigmoid/16
// ---------------------------------------------------------------------------
__global__ __launch_bounds__(256) void softmax64(const float* __restrict__ X,
                                                 float* __restrict__ Y, size_t nrows) {
  size_t row = (size_t)blockIdx.x * blockDim.x + threadIdx.x;
  if (row >= nrows) return;
  const float* x = X + row * 64;
  float* y = Y + row * 64;
  float mx = -3e38f;
  for (int d = 0; d < 64; d++) mx = fmaxf(mx, x[d]);
  float s = 0.0f;
  for (int d = 0; d < 64; d++) s += __expf(x[d] - mx);
  float inv = 1.0f / s;
  for (int d = 0; d < 64; d++) y[d] = __expf(x[d] - mx) * inv;
}

__global__ __launch_bounds__(256) void gate_logsig(const float* __restrict__ X,
                                                   float* __restrict__ Y, size_t n) {
  size_t i = (size_t)blockIdx.x * blockDim.x + threadIdx.x;
  if (i >= n) return;
  float x = X[i];
  float ls = (x > 0.0f) ? -log1pf(__expf(-x)) : (x - log1pf(__expf(x)));
  Y[i] = ls * (1.0f / 16.0f);
}

// ---------------------------------------------------------------------------
// Chunked gated linear attention (student). grid = B*NKV blocks, block 256.
// ---------------------------------------------------------------------------
__global__ __launch_bounds__(256) void gla_chunk(const float* __restrict__ Qsm,
                                                 const float* __restrict__ Ksm,
                                                 const float* __restrict__ Vf,
                                                 const float* __restrict__ G,
                                                 float* __restrict__ O) {
  __shared__ __attribute__((aligned(16))) char R0raw[16384];
  __shared__ __attribute__((aligned(16))) bf16 EQ[64][64];  // exp(Gcum) bf16
  __shared__ __attribute__((aligned(16))) bf16 KT[64][64];  // k~ [t][d]
  __shared__ __attribute__((aligned(16))) bf16 KH[64][64];  // k^ [d][t]
  __shared__ __attribute__((aligned(16))) bf16 VT[64][64];  // v  [dv][t]
  __shared__ __attribute__((aligned(16))) bf16 SB[64][64];  // state [v][d]
  __shared__ float GEND[64];
  float (*GC)[64] = (float (*)[64])R0raw;
  bf16 (*QH)[64]  = (bf16 (*)[64])R0raw;
  bf16 (*AB)[64]  = (bf16 (*)[64])(R0raw + 8192);

  int kvh = blockIdx.x & 7, b = blockIdx.x >> 3;
  int tid = threadIdx.x, lane = tid & 31, wave = tid >> 5;
  int hi8 = (lane & 16) >> 1;
  int col = lane & 15;
  int tload = tid >> 2, d0 = (tid & 3) * 16;

  v8f sreg[2];
  sreg[0] = vzero8();
  sreg[1] = vzero8();
  for (int i = tid; i < 64 * 64; i += 256) ((bf16*)SB)[i] = (bf16)0.0f;

  for (int cc = 0; cc < SS / 64; cc++) {
    int sbase = cc * 64;
    __syncthreads();
    {
      const float* gp = G + ((size_t)(b * SS + sbase + tload) * NKVH + kvh) * HDIM + d0;
#pragma unroll
      for (int j = 0; j < 16; j++) GC[tload][d0 + j] = gp[j];
    }
    __syncthreads();
    if (tid < 64) {
      float run = 0.0f;
      for (int t = 0; t < 64; t++) { run += GC[t][tid]; GC[t][tid] = run; }
      GEND[tid] = run;
    }
    __syncthreads();
    {
      const float* kp = Ksm + ((size_t)(b * SS + sbase + tload) * NKVH + kvh) * HDIM + d0;
      const float* vp = Vf  + ((size_t)(b * SS + sbase + tload) * NKVH + kvh) * HDIM + d0;
#pragma unroll
      for (int j = 0; j < 16; j++) {
        int d = d0 + j;
        float Gtd = GC[tload][d];
        float kv = kp[j];
        EQ[tload][d] = (bf16)__expf(Gtd);
        KT[tload][d] = (bf16)(kv * __expf(-Gtd));
        KH[d][tload] = (bf16)(kv * __expf(GEND[d] - Gtd));
        VT[d][tload] = (bf16)vp[j];
      }
    }
    __syncthreads();
    const float* qbase =
        Qsm + ((size_t)(b * SS + sbase + tload) * NHEAD + kvh * 4) * HDIM + d0;
    for (int hh = 0; hh < 4; hh++) {
#pragma unroll
      for (int j = 0; j < 16; j++) {
        int d = d0 + j;
        QH[tload][d] = (bf16)(qbase[hh * HDIM + j] * (float)EQ[tload][d]);
      }
      __syncthreads();
#pragma unroll
      for (int i = 0; i < 2; i++) {
        int id = wave * 2 + i;
        int tq = id >> 2, tk = id & 3;
        if (tk > tq) {
#pragma unroll
          for (int r = 0; r < 8; r++) AB[tq * 16 + r + hi8][tk * 16 + col] = (bf16)0.0f;
        } else {
          v8f acc = vzero8();
          const bf16* ar = &QH[tq * 16 + col][0];
          const bf16* br = &KT[tk * 16 + col][0];
#pragma unroll
          for (int kk = 0; kk < 64; kk += 32)
            acc = WMMA_BF(fragA(ar, kk, lane), fragB(br, kk, lane), acc);
#pragma unroll
          for (int r = 0; r < 8; r++) {
            int rl = r + hi8;
            float v = acc[r];
            if (tq == tk && col > rl) v = 0.0f;
            AB[tq * 16 + rl][tk * 16 + col] = (bf16)v;
          }
        }
      }
      __syncthreads();
#pragma unroll
      for (int i = 0; i < 2; i++) {
        int id = wave * 2 + i;
        int tq = id >> 2, dv = id & 3;
        v8f acc = vzero8();
        const bf16* aq = &QH[tq * 16 + col][0];
        const bf16* bs = &SB[dv * 16 + col][0];
        const bf16* aa = &AB[tq * 16 + col][0];
        const bf16* bv = &VT[dv * 16 + col][0];
#pragma unroll
        for (int kk = 0; kk < 64; kk += 32) {
          acc = WMMA_BF(fragA(aq, kk, lane), fragB(bs, kk, lane), acc);
          acc = WMMA_BF(fragA(aa, kk, lane), fragB(bv, kk, lane), acc);
        }
#pragma unroll
        for (int r = 0; r < 8; r++) {
          int srow = sbase + tq * 16 + r + hi8;
          O[((size_t)(b * SS + srow) * NHEAD + kvh * 4 + hh) * HDIM + dv * 16 + col] =
              acc[r];
        }
      }
      __syncthreads();
    }
#pragma unroll
    for (int i = 0; i < 2; i++) {
      int id = wave * 2 + i;
      int td = id >> 2, tv = id & 3;
      v8f acc = vzero8();
      const bf16* ak = &KH[td * 16 + col][0];
      const bf16* bv = &VT[tv * 16 + col][0];
#pragma unroll
      for (int kk = 0; kk < 64; kk += 32)
        acc = WMMA_BF(fragA(ak, kk, lane), fragB(bv, kk, lane), acc);
#pragma unroll
      for (int r = 0; r < 8; r++) {
        int d = td * 16 + r + hi8;
        int v = tv * 16 + col;
        float nv = __expf(GEND[d]) * sreg[i][r] + acc[r];
        sreg[i][r] = nv;
        SB[v][d] = (bf16)nv;
      }
    }
  }
}

// ---------------------------------------------------------------------------
// Host launcher
// ---------------------------------------------------------------------------
extern "C" void kernel_launch(void* const* d_in, const int* in_sizes, int n_in,
                              void* d_out, int out_size, void* d_ws, size_t ws_size,
                              hipStream_t stream) {
  const float* hs  = (const float*)d_in[0];
  const float* wq  = (const float*)d_in[1];
  const float* wk  = (const float*)d_in[2];
  const float* wv  = (const float*)d_in[3];
  const float* wo  = (const float*)d_in[4];
  const float* wqs = (const float*)d_in[5];
  const float* wks = (const float*)d_in[6];
  const float* wvs = (const float*)d_in[7];
  const float* wos = (const float*)d_in[8];
  float* out_t = (float*)d_out;
  float* out_s = out_t + (size_t)MROWS * HHID;

  size_t off = 0;
  auto take = [&](size_t bytes) -> void* {
    void* r = (char*)d_ws + off;
    off = (off + bytes + 255) & ~(size_t)255;
    return r;
  };
  bf16* hs_bf  = (bf16*)take((size_t)MROWS * HHID * 2);
  bf16* wq_bf  = (bf16*)take((size_t)DQ * HHID * 2);
  bf16* wk_bf  = (bf16*)take((size_t)DKV * HHID * 2);
  bf16* wv_bf  = (bf16*)take((size_t)DKV * HHID * 2);
  bf16* wo_bf  = (bf16*)take((size_t)HHID * DQ * 2);
  bf16* wqs_bf = (bf16*)take((size_t)DQ * HHID * 2);
  bf16* wks_bf = (bf16*)take((size_t)DKV * HHID * 2);
  bf16* wvs_bf = (bf16*)take((size_t)DKV * HHID * 2);
  bf16* wos_bf = (bf16*)take((size_t)HHID * DQ * 2);
  float* q_t  = (float*)take((size_t)MROWS * DQ * 4);
  float* k_t  = (float*)take((size_t)MROWS * DKV * 4);
  float* v_t  = (float*)take((size_t)MROWS * DKV * 4);
  bf16* qro   = (bf16*)take((size_t)MROWS * DQ * 2);
  bf16* kro   = (bf16*)take((size_t)MROWS * DKV * 2);
  bf16* vtr   = (bf16*)take((size_t)MROWS * DKV * 2);
  float* o_t  = (float*)take((size_t)MROWS * DQ * 4);
  float* q_s  = (float*)take((size_t)MROWS * DQ * 4);
  float* k_s  = (float*)take((size_t)MROWS * DKV * 4);
  float* v_s  = (float*)take((size_t)MROWS * DKV * 4);
  float* q_sm = (float*)take((size_t)MROWS * DQ * 4);
  float* k_sm = (float*)take((size_t)MROWS * DKV * 4);
  float* gbuf = (float*)take((size_t)MROWS * DKV * 4);
  float* o_s  = (float*)take((size_t)MROWS * DQ * 4);
  bf16* ot_bf = (bf16*)take((size_t)MROWS * DQ * 2);
  bf16* os_bf = (bf16*)take((size_t)MROWS * DQ * 2);

  auto cvt = [&](const float* s, bf16* d, size_t n) {
    cvt_f32_bf16<<<dim3((unsigned)((n + 255) / 256)), dim3(256), 0, stream>>>(s, d, n);
  };

  cvt(hs, hs_bf, (size_t)MROWS * HHID);
  cvt(wq, wq_bf, (size_t)DQ * HHID);
  cvt(wk, wk_bf, (size_t)DKV * HHID);
  cvt(wv, wv_bf, (size_t)DKV * HHID);
  cvt(wo, wo_bf, (size_t)HHID * DQ);
  cvt(wqs, wqs_bf, (size_t)DQ * HHID);
  cvt(wks, wks_bf, (size_t)DKV * HHID);
  cvt(wvs, wvs_bf, (size_t)DKV * HHID);
  cvt(wos, wos_bf, (size_t)HHID * DQ);

  gemm_bf16_wmma<<<dim3(MROWS / 128, DQ / 64), dim3(256), 0, stream>>>(hs_bf, wq_bf, q_t, HHID, DQ);
  gemm_bf16_wmma<<<dim3(MROWS / 128, DKV / 64), dim3(256), 0, stream>>>(hs_bf, wk_bf, k_t, HHID, DKV);
  gemm_bf16_wmma<<<dim3(MROWS / 128, DKV / 64), dim3(256), 0, stream>>>(hs_bf, wv_bf, v_t, HHID, DKV);
  gemm_bf16_wmma<<<dim3(MROWS / 128, DQ / 64), dim3(256), 0, stream>>>(hs_bf, wqs_bf, q_s, HHID, DQ);
  gemm_bf16_wmma<<<dim3(MROWS / 128, DKV / 64), dim3(256), 0, stream>>>(hs_bf, wks_bf, k_s, HHID, DKV);
  gemm_bf16_wmma<<<dim3(MROWS / 128, DKV / 64), dim3(256), 0, stream>>>(hs_bf, wvs_bf, v_s, HHID, DKV);

  {
    size_t nq = (size_t)BB * NHEAD * SS * 32;
    rope_pack<<<dim3((unsigned)((nq + 255) / 256)), dim3(256), 0, stream>>>(q_t, qro, NHEAD);
    size_t nk = (size_t)BB * NKVH * SS * 32;
    rope_pack<<<dim3((unsigned)((nk + 255) / 256)), dim3(256), 0, stream>>>(k_t, kro, NKVH);
    size_t nv = (size_t)BB * NKVH * SS * HDIM;
    headmajor_cvt<<<dim3((unsigned)((nv + 255) / 256)), dim3(256), 0, stream>>>(v_t, vtr);
  }
  flash_attn<<<dim3(SS / 64, NHEAD, BB), dim3(256), 0, stream>>>(qro, kro, vtr, o_t);

  softmax64<<<dim3((MROWS * NHEAD) / 256), dim3(256), 0, stream>>>(q_s, q_sm, (size_t)MROWS * NHEAD);
  softmax64<<<dim3((MROWS * NKVH) / 256), dim3(256), 0, stream>>>(k_s, k_sm, (size_t)MROWS * NKVH);
  gate_logsig<<<dim3((unsigned)(((size_t)MROWS * DKV) / 256)), dim3(256), 0, stream>>>(
      k_s, gbuf, (size_t)MROWS * DKV);
  gla_chunk<<<dim3(BB * NKVH), dim3(256), 0, stream>>>(q_sm, k_sm, v_s, gbuf, o_s);

  cvt(o_t, ot_bf, (size_t)MROWS * DQ);
  cvt(o_s, os_bf, (size_t)MROWS * DQ);
  gemm_bf16_wmma<<<dim3(MROWS / 128, HHID / 64), dim3(256), 0, stream>>>(ot_bf, wo_bf, out_t, DQ, HHID);
  gemm_bf16_wmma<<<dim3(MROWS / 128, HHID / 64), dim3(256), 0, stream>>>(os_bf, wos_bf, out_s, DQ, HHID);
}
